// MatchLayer_48567490183531
// MI455X (gfx1250) — matrix operand order repaired and checked by reading, not compiled, not run on hardware
//
#include <hip/hip_runtime.h>

typedef __bf16 v16bf __attribute__((ext_vector_type(16)));
typedef __bf16 v8bf  __attribute__((ext_vector_type(8)));
typedef float  v8f   __attribute__((ext_vector_type(8)));

#define BATCH 32
#define SEQ   1024
#define DIM   512
#define BS    (BATCH * SEQ)
#define EPSV  1e-6f

// ===========================================================================
// Kernel A: fused fp32->bf16 conversion + reciprocal L2 norms.
// One wave32 per 512-float row (lane owns 16 contiguous elements).
// wnorm layout (floats): [0..BS) rtn_fwd, [BS..2BS) rtn_bwd,
//                        [2BS..3BS) rhn_fwd, [3BS..4BS) rhn_bwd
// wbf layout (__bf16):   4 arrays of [BS, DIM], same order as inputs.
// ===========================================================================
__global__ __launch_bounds__(256)
void convert_norms_kernel(const float* __restrict__ tf, const float* __restrict__ tb,
                          const float* __restrict__ hf, const float* __restrict__ hb,
                          float* __restrict__ wnorm, __bf16* __restrict__ wbf) {
    const int rowGlobal = blockIdx.x * 8 + (threadIdx.x >> 5); // 0 .. 4*BS-1
    const int lane = threadIdx.x & 31;
    const int arr  = rowGlobal >> 15;          // / BS (BS = 32768)
    const int row  = rowGlobal & (BS - 1);

    const float* src = (arr == 0) ? tf : (arr == 1) ? tb : (arr == 2) ? hf : hb;
    const float* p = src + (size_t)row * DIM + lane * 16;
    __bf16* q = wbf + ((size_t)arr * BS + row) * DIM + lane * 16;

    float ss = 0.0f;
    v16bf o;
#pragma unroll
    for (int t = 0; t < 4; ++t) {
        const float4 v = *(const float4*)(p + 4 * t);
        ss += v.x * v.x + v.y * v.y + v.z * v.z + v.w * v.w;
        o[4 * t + 0] = (__bf16)v.x;
        o[4 * t + 1] = (__bf16)v.y;
        o[4 * t + 2] = (__bf16)v.z;
        o[4 * t + 3] = (__bf16)v.w;
    }
    *(v16bf*)q = o;

#pragma unroll
    for (int m = 16; m >= 1; m >>= 1) ss += __shfl_xor(ss, m, 32);
    if (lane == 0) wnorm[rowGlobal] = 1.0f / sqrtf(fmaxf(ss, EPSV));
}

// Norms-only variant (fallback path when ws is too small for bf16 copies).
__global__ __launch_bounds__(256)
void norms_kernel(const float* __restrict__ tf, const float* __restrict__ tb,
                  const float* __restrict__ hf, const float* __restrict__ hb,
                  float* __restrict__ wnorm) {
    const int rowGlobal = blockIdx.x * 8 + (threadIdx.x >> 5);
    const int lane = threadIdx.x & 31;
    const int arr  = rowGlobal >> 15;
    const int row  = rowGlobal & (BS - 1);
    const float* src = (arr == 0) ? tf : (arr == 1) ? tb : (arr == 2) ? hf : hb;
    const float* p = src + (size_t)row * DIM;

    float ss = 0.0f;
#pragma unroll
    for (int q = 0; q < 4; ++q) {
        const float4 v = *(const float4*)(p + q * 128 + lane * 4);
        ss += v.x * v.x + v.y * v.y + v.z * v.z + v.w * v.w;
    }
#pragma unroll
    for (int m = 16; m >= 1; m >>= 1) ss += __shfl_xor(ss, m, 32);
    if (lane == 0) wnorm[rowGlobal] = 1.0f / sqrtf(fmaxf(ss, EPSV));
}

// ===========================================================================
// WMMA fragment loaders (wave32; this code's contiguous k = 16h + s order, same for A and B).
// A (16x32 bf16): lane = M (mod 16); low half-lanes K 0..15,
//                 high half-lanes K 16..31.
// B (32x16 bf16): lane = N (mod 16); low half-lanes K 0..15, high K 16..31.
// ===========================================================================
__device__ __forceinline__ v16bf load_fragA_bf(const __bf16* __restrict__ rowPtr,
                                               int k0, int lane) {
    const __bf16* p0 = rowPtr + k0 + ((lane & 16) ? 16 : 0);
    const v8bf lo = *(const v8bf*)(p0);        // 16B aligned
    const v8bf hi = *(const v8bf*)(p0 + 8);    // 16B aligned
    return __builtin_shufflevector(lo, hi, 0, 1, 2, 3, 4, 5, 6, 7,
                                           8, 9, 10, 11, 12, 13, 14, 15);
}

__device__ __forceinline__ v16bf load_fragB_bf(const __bf16* __restrict__ rowPtr,
                                               int k0, int lane) {
    return *(const v16bf*)(rowPtr + k0 + ((lane & 16) ? 16 : 0)); // 32B aligned
}

// fp32-source variants (fallback path: convert in-loop)
__device__ __forceinline__ v16bf load_fragA_f32(const float* __restrict__ rowPtr,
                                                int k0, int lane) {
    const float* p0 = rowPtr + k0 + ((lane & 16) ? 16 : 0);
    v16bf r;
#pragma unroll
    for (int t = 0; t < 8; ++t) r[t] = (__bf16)p0[t];
#pragma unroll
    for (int t = 0; t < 8; ++t) r[8 + t] = (__bf16)p0[8 + t];
    return r;
}
__device__ __forceinline__ v16bf load_fragB_f32(const float* __restrict__ rowPtr,
                                                int k0, int lane) {
    const float* p0 = rowPtr + k0 + ((lane & 16) ? 16 : 0);
    v16bf r;
#pragma unroll
    for (int t = 0; t < 16; ++t) r[t] = (__bf16)p0[t];
    return r;
}

// ===========================================================================
// Shared epilogue: scale by 1/||text_j||, fold max/sum over columns,
// 16-lane butterfly, cross-wave LDS combine, scale by 1/||hypo_i||.
// ===========================================================================
__device__ __forceinline__
void pool_epilogue(v8f c[8], const float* __restrict__ rt,
                   const float* __restrict__ rh,
                   float* __restrict__ out_max, float* __restrict__ out_mean,
                   int b, int i0, int wave, int lane, int laneRow) {
    float rmax[8], rsum[8];
#pragma unroll
    for (int v = 0; v < 8; ++v) { rmax[v] = -3.402823466e38f; rsum[v] = 0.0f; }
#pragma unroll
    for (int t = 0; t < 8; ++t) {
        const float s = rt[(wave + 8 * t) * 16 + laneRow];
#pragma unroll
        for (int v = 0; v < 8; ++v) {
            const float val = c[t][v] * s;
            rmax[v] = fmaxf(rmax[v], val);
            rsum[v] += val;
        }
    }
#pragma unroll
    for (int m = 1; m < 16; m <<= 1) {
#pragma unroll
        for (int v = 0; v < 8; ++v) {
            rmax[v] = fmaxf(rmax[v], __shfl_xor(rmax[v], m, 32));
            rsum[v] += __shfl_xor(rsum[v], m, 32);
        }
    }
    __shared__ float lmax[8][16];
    __shared__ float lsum[8][16];
    if (laneRow == 0) {
        const int rbase = (lane & 16) ? 8 : 0;
#pragma unroll
        for (int v = 0; v < 8; ++v) {
            lmax[wave][rbase + v] = rmax[v];
            lsum[wave][rbase + v] = rsum[v];
        }
    }
    __syncthreads();
    if (threadIdx.x < 16) {
        const int r = threadIdx.x;
        float m = lmax[0][r];
        float s = lsum[0][r];
#pragma unroll
        for (int w = 1; w < 8; ++w) {
            m = fmaxf(m, lmax[w][r]);
            s += lsum[w][r];
        }
        const float rhv = rh[i0 + r];
        const size_t o = (size_t)b * SEQ + i0 + r;
        out_max[o]  = m * rhv;
        out_mean[o] = s * rhv * (1.0f / (float)SEQ);
    }
}

// ===========================================================================
// Kernel B (main path): bf16-native fused GEMM + pooling, software-pipelined.
// grid.x = SEQ/16 (hypo row tile), grid.y = BATCH; 8 waves/block.
// Wave owns 8 N-tiles: n0 = (wave + 8*t)*16. Depth-2 B prefetch ring keeps
// ~4 b128 loads in flight across each WMMA; A fragment prefetched one k ahead.
// ===========================================================================
__global__ __launch_bounds__(256)
void matchpool_bf16(const __bf16* __restrict__ text,
                    const __bf16* __restrict__ hypo,
                    const float* __restrict__ rtn, const float* __restrict__ rhn,
                    float* __restrict__ out_max, float* __restrict__ out_mean) {
    const int itile = blockIdx.x;
    const int b     = blockIdx.y;
    const int wave  = threadIdx.x >> 5;
    const int lane  = threadIdx.x & 31;
    const int i0    = itile * 16;
    const int laneRow = lane & 15;

    const __bf16* aRow  = hypo + ((size_t)b * SEQ + i0 + laneRow) * DIM;
    const __bf16* tbase = text + (size_t)b * SEQ * DIM;
    const float* rt = rtn + b * SEQ;
    const float* rh = rhn + b * SEQ;

    const __bf16* bRow[8];
#pragma unroll
    for (int t = 0; t < 8; ++t)
        bRow[t] = tbase + (size_t)((wave + 8 * t) * 16 + laneRow) * DIM;

    v8f c[8];
#pragma unroll
    for (int t = 0; t < 8; ++t)
#pragma unroll
        for (int v = 0; v < 8; ++v) c[t][v] = 0.0f;

    // Prime the pipeline: A for k=0, B tiles 0 and 1 for k=0.
    v16bf a  = load_fragA_bf(aRow, 0, lane);
    v16bf b0 = load_fragB_bf(bRow[0], 0, lane);
    v16bf b1 = load_fragB_bf(bRow[1], 0, lane);

    // Steady state: branch-free, two B tiles in flight + next-k A in flight.
    for (int k0 = 0; k0 < DIM - 32; k0 += 32) {
        const v16bf anext = load_fragA_bf(aRow, k0 + 32, lane);
#pragma unroll
        for (int t = 0; t < 8; ++t) {
            const v16bf bpre = (t < 6)
                ? load_fragB_bf(bRow[t + 2], k0, lane)        // tiles t+2 of k0
                : load_fragB_bf(bRow[t - 6], k0 + 32, lane);  // tiles 0,1 of k0+32
            c[t] = __builtin_amdgcn_wmma_f32_16x16x32_bf16(
                false, a, false, b0, (short)0, c[t], false, false);
            b0 = b1;
            b1 = bpre;
        }
        a = anext;
    }
    // Peeled final k-step (k0 = DIM-32): drain the ring.
#pragma unroll
    for (int t = 0; t < 8; ++t) {
        v16bf bpre = b1;
        if (t < 6) bpre = load_fragB_bf(bRow[t + 2], DIM - 32, lane);
        c[t] = __builtin_amdgcn_wmma_f32_16x16x32_bf16(
            false, a, false, b0, (short)0, c[t], false, false);
        b0 = b1;
        b1 = bpre;
    }

    pool_epilogue(c, rt, rh, out_max, out_mean, b, i0, wave, lane, laneRow);
}

// Fallback: fp32 source, convert inside the loop (round-1 scheme).
__global__ __launch_bounds__(256)
void matchpool_f32(const float* __restrict__ text,
                   const float* __restrict__ hypo,
                   const float* __restrict__ rtn, const float* __restrict__ rhn,
                   float* __restrict__ out_max, float* __restrict__ out_mean) {
    const int itile = blockIdx.x;
    const int b     = blockIdx.y;
    const int wave  = threadIdx.x >> 5;
    const int lane  = threadIdx.x & 31;
    const int i0    = itile * 16;
    const int laneRow = lane & 15;

    const float* aRow  = hypo + ((size_t)b * SEQ + i0 + laneRow) * DIM;
    const float* tbase = text + (size_t)b * SEQ * DIM;
    const float* rt = rtn + b * SEQ;
    const float* rh = rhn + b * SEQ;

    const float* bRow[8];
#pragma unroll
    for (int t = 0; t < 8; ++t)
        bRow[t] = tbase + (size_t)((wave + 8 * t) * 16 + laneRow) * DIM;

    v8f c[8];
#pragma unroll
    for (int t = 0; t < 8; ++t)
#pragma unroll
        for (int v = 0; v < 8; ++v) c[t][v] = 0.0f;

    for (int k0 = 0; k0 < DIM; k0 += 32) {
        const v16bf a = load_fragA_f32(aRow, k0, lane);
#pragma unroll
        for (int t = 0; t < 8; ++t) {
            const v16bf bf = load_fragB_f32(bRow[t], k0, lane);
            c[t] = __builtin_amdgcn_wmma_f32_16x16x32_bf16(
                false, a, false, bf, (short)0, c[t], false, false);
        }
    }
    pool_epilogue(c, rt, rh, out_max, out_mean, b, i0, wave, lane, laneRow);
}

// ===========================================================================
extern "C" void kernel_launch(void* const* d_in, const int* in_sizes, int n_in,
                              void* d_out, int out_size, void* d_ws, size_t ws_size,
                              hipStream_t stream) {
    const float* tf = (const float*)d_in[0];  // text_fwd  [B,S,D]
    const float* tb = (const float*)d_in[1];  // text_bwd
    const float* hf = (const float*)d_in[2];  // hypo_fwd
    const float* hb = (const float*)d_in[3];  // hypo_bwd
    float* out = (float*)d_out;               // 4 x [B*S] concatenated
    float* wnorm = (float*)d_ws;              // 4 x [B*S] reciprocal norms

    const size_t normBytes = (size_t)4 * BS * sizeof(float);
    const size_t bfBytes   = (size_t)4 * BS * DIM * sizeof(__bf16);
    const dim3 grid(SEQ / 16, BATCH);

    if (ws_size >= normBytes + bfBytes) {
        // Main path: one-pass convert+norms, then bf16-native pipelined GEMM.
        __bf16* wbf = (__bf16*)((char*)d_ws + normBytes);
        convert_norms_kernel<<<(4 * BS) / 8, 256, 0, stream>>>(tf, tb, hf, hb,
                                                               wnorm, wbf);
        const size_t AS = (size_t)BS * DIM;
        matchpool_bf16<<<grid, 256, 0, stream>>>(
            wbf + 0 * AS, wbf + 2 * AS, wnorm + 0 * BS, wnorm + 2 * BS,
            out + 0 * BS, out + 1 * BS);
        matchpool_bf16<<<grid, 256, 0, stream>>>(
            wbf + 1 * AS, wbf + 3 * AS, wnorm + 1 * BS, wnorm + 3 * BS,
            out + 2 * BS, out + 3 * BS);
    } else {
        // Fallback: norms only in ws; convert fp32->bf16 inside the GEMM.
        norms_kernel<<<(4 * BS) / 8, 256, 0, stream>>>(tf, tb, hf, hb, wnorm);
        matchpool_f32<<<grid, 256, 0, stream>>>(
            tf, hf, wnorm + 0 * BS, wnorm + 2 * BS, out + 0 * BS, out + 1 * BS);
        matchpool_f32<<<grid, 256, 0, stream>>>(
            tb, hb, wnorm + 1 * BS, wnorm + 3 * BS, out + 2 * BS, out + 3 * BS);
    }
}
